// QKVAttention_2551210574387
// MI455X (gfx1250) — matrix operand order repaired and checked by reading, not compiled, not run on hardware
//
#include <hip/hip_runtime.h>

// MI455X / gfx1250: wave32, WMMA 16x16x32 bf16 -> f32 accumulate.
// Attention pipeline: qkv proj (GEMM) -> flash attention -> out proj + residual.
// Round 2: async-to-LDS staging of K tiles (ASYNCcnt path) + global prefetch.

typedef __attribute__((ext_vector_type(16))) __bf16 v16bf;
typedef __attribute__((ext_vector_type(8)))  __bf16 v8bf;
typedef __attribute__((ext_vector_type(8)))  float  v8f;

#define B_N 8
#define C_N 512
#define T_N 2048
#define SOFT_SCALE 0.044194173824159216f  // 1/sqrt(512)

static __device__ __forceinline__ __bf16 f2bf(float f) {
  unsigned u = __builtin_bit_cast(unsigned, f);
  unsigned r = u + 0x7FFFu + ((u >> 16) & 1u);
  unsigned short h = (unsigned short)(r >> 16);
  return __builtin_bit_cast(__bf16, h);
}

static __device__ __forceinline__ v8f wmma_bf16(v16bf a, v16bf b, v8f c) {
  // D = A(16x32) x B(32x16) + C, fp32 accumulate
  return __builtin_amdgcn_wmma_f32_16x16x32_bf16(false, a, false, b, (short)0, c,
                                                 false, false);
}

// Load a 16x32 bf16 fragment (A layout; B uses the same per-lane K split with
// n = lane&15). Rows are contiguous with `rowStride` elements between rows.
// Per ISA 7.12.2: lanes 0-15 hold K={0..7,16..23}, lanes 16-31 hold K={8..15,24..31}.
static __device__ __forceinline__ v16bf load_frag(const __bf16* row0, long rowStride,
                                                  int lane) {
  int m = lane & 15;
  int ko = (lane & 16) ? 8 : 0;
  const __bf16* p = row0 + (long)m * rowStride;
  v8bf lo = *(const v8bf*)(p + ko);
  v8bf hi = *(const v8bf*)(p + 16 + ko);
  return __builtin_shufflevector(lo, hi, 0, 1, 2, 3, 4, 5, 6, 7, 8, 9, 10, 11, 12,
                                 13, 14, 15);
}

// Async global->LDS copy of 16 bytes (CDNA5 GLOBAL_LOAD_ASYNC_TO_LDS_B128,
// GV mode, tracked by ASYNCcnt).  Low 32 bits of a generic LDS pointer are the
// wave-relative LDS byte address.
static __device__ __forceinline__ void async_ld16(void* lds, const void* gptr) {
  unsigned loff = (unsigned)(unsigned long long)lds;
  asm volatile("global_load_async_to_lds_b128 %0, %1, off"
               :
               : "v"(loff), "v"(gptr)
               : "memory");
}
static __device__ __forceinline__ void wait_async0() {
  asm volatile("s_wait_asynccnt 0" ::: "memory");
}

// ---------------------------------------------------------------------------
// Kernel 1: q/k/v projections.  For each batch b and proj p:
//   P[o,t] = sum_c W[o,c] * x[b,c,t] + bias[o]
// Q,K stored transposed as [B,T,C] bf16 ; V stored natural [B,C,T] bf16.
// Block = 256 thr (8 waves), tile 128x128, K-step 32.
// ---------------------------------------------------------------------------
__global__ __launch_bounds__(256) void qkv_proj_kernel(
    const float* __restrict__ x, const float* __restrict__ Wq,
    const float* __restrict__ bq, const float* __restrict__ Wk,
    const float* __restrict__ bk, const float* __restrict__ Wv,
    const float* __restrict__ bv, __bf16* __restrict__ Qt,
    __bf16* __restrict__ Kt, __bf16* __restrict__ V) {
  __shared__ __align__(16) __bf16 Al[128 * 40];  // [m][k], stride 40 (pad)
  __shared__ __align__(16) __bf16 Bl[128 * 40];  // [n][k] (x transposed)

  const int tid = threadIdx.x;
  const int lane = tid & 31;
  const int wave = tid >> 5;
  const int waveM = wave & 1;   // 2 x 64 rows
  const int waveN = wave >> 1;  // 4 x 32 cols

  const int o0 = blockIdx.x * 128;
  const int t0 = blockIdx.y * 128;
  const int z = blockIdx.z;
  const int b = z / 3;
  const int which = z - 3 * b;

  const float* W = (which == 0) ? Wq : (which == 1) ? Wk : Wv;
  const float* bias = (which == 0) ? bq : (which == 1) ? bk : bv;

  v8f acc[4][2];
#pragma unroll
  for (int i = 0; i < 4; ++i)
#pragma unroll
    for (int j = 0; j < 2; ++j) acc[i][j] = (v8f)0.0f;

  for (int kc = 0; kc < C_N; kc += 32) {
    // A tile: W[o0+m][kc..kc+31], fp32 -> bf16
    {
      int m = tid >> 1;
      int k0 = (tid & 1) * 16;
      const float* src = W + (long)(o0 + m) * C_N + kc + k0;
#pragma unroll
      for (int j = 0; j < 2; ++j) {
        float4 fa = *(const float4*)(src + j * 8);
        float4 fb = *(const float4*)(src + j * 8 + 4);
        v8bf t;
        t[0] = f2bf(fa.x); t[1] = f2bf(fa.y); t[2] = f2bf(fa.z); t[3] = f2bf(fa.w);
        t[4] = f2bf(fb.x); t[5] = f2bf(fb.y); t[6] = f2bf(fb.z); t[7] = f2bf(fb.w);
        *(v8bf*)&Al[m * 40 + k0 + j * 8] = t;
      }
      if (kc + 32 < C_N) __builtin_prefetch(src + 32, 0, 0);  // next A tile
    }
    // B tile (transposed into [n][k]): Bl[n][k] = x[b][kc+k][t0+n]
    {
#pragma unroll
      for (int kk = 0; kk < 4; ++kk) {
        int k = kk * 8 + wave;
        int n0 = (tid & 31) * 4;
        const float* src = x + ((long)b * C_N + kc + k) * T_N + t0 + n0;
        float4 f = *(const float4*)src;
        Bl[(n0 + 0) * 40 + k] = f2bf(f.x);
        Bl[(n0 + 1) * 40 + k] = f2bf(f.y);
        Bl[(n0 + 2) * 40 + k] = f2bf(f.z);
        Bl[(n0 + 3) * 40 + k] = f2bf(f.w);
        if (kc + 32 < C_N) __builtin_prefetch(src + 32 * T_N, 0, 0);  // next B tile
      }
    }
    __syncthreads();
    v16bf af[4], bfr[2];
#pragma unroll
    for (int mt = 0; mt < 4; ++mt)
      af[mt] = load_frag(&Al[(waveM * 64 + mt * 16) * 40], 40, lane);
#pragma unroll
    for (int nt = 0; nt < 2; ++nt)
      bfr[nt] = load_frag(&Bl[(waveN * 32 + nt * 16) * 40], 40, lane);
#pragma unroll
    for (int mt = 0; mt < 4; ++mt)
#pragma unroll
      for (int nt = 0; nt < 2; ++nt)
        acc[mt][nt] = wmma_bf16(af[mt], bfr[nt], acc[mt][nt]);
    __syncthreads();
  }

  // Epilogue: add bias, convert, store per layout.
  const int n_l = lane & 15;
  const int hi8 = (lane >> 4) << 3;
#pragma unroll
  for (int mt = 0; mt < 4; ++mt) {
    int ob = o0 + waveM * 64 + mt * 16 + hi8;
#pragma unroll
    for (int nt = 0; nt < 2; ++nt) {
      int tg = t0 + waveN * 32 + nt * 16 + n_l;
      if (which == 2) {  // V: [B,C,T]
#pragma unroll
        for (int r = 0; r < 8; ++r)
          V[((long)b * C_N + ob + r) * T_N + tg] = f2bf(acc[mt][nt][r] + bias[ob + r]);
      } else {  // Q/K: [B,T,C]
        __bf16* dst = ((which == 0) ? Qt : Kt) + ((long)b * T_N + tg) * C_N + ob;
        v8bf t;
#pragma unroll
        for (int r = 0; r < 8; ++r) t[r] = f2bf(acc[mt][nt][r] + bias[ob + r]);
        *(v8bf*)dst = t;
      }
    }
  }
}

// ---------------------------------------------------------------------------
// Kernel 2: flash attention.  One (b, 32-row t-tile) per workgroup.
// K tiles are staged into LDS with GLOBAL_LOAD_ASYNC_TO_LDS_B128 (waves w and
// w+4 consume identical K fragments, so LDS staging halves L2 traffic in the
// QK^T loop).  Online softmax; O += P @ V^T (V fragments direct from global,
// unique per wave).
// ---------------------------------------------------------------------------
__global__ __launch_bounds__(256) void flash_attn_kernel(
    const __bf16* __restrict__ Qt, const __bf16* __restrict__ Kt,
    const __bf16* __restrict__ V, __bf16* __restrict__ Obf) {
  __shared__ __align__(16) __bf16 Klds[128 * 72];  // 128 s-rows x 64 c (pad 72)
  __shared__ __align__(16) float Sl[32 * 132];     // scores tile
  __shared__ __align__(16) __bf16 Pl[32 * 136];    // softmaxed probs (bf16)
  __shared__ float rowM[32], rowL[32], rowScale[32];

  const int tid = threadIdx.x;
  const int lane = tid & 31;
  const int wave = tid >> 5;
  const int t0 = blockIdx.x * 32;
  const int b = blockIdx.y;

  const int n_l = lane & 15;
  const int hi8 = (lane >> 4) << 3;

  const int sMt = wave >> 2;        // score M-tile (0..1)
  const int sNg = (wave & 3) * 32;  // score col group (2 N-tiles)

  v8f accO[2][4];
#pragma unroll
  for (int mt = 0; mt < 2; ++mt)
#pragma unroll
    for (int nt = 0; nt < 4; ++nt) accO[mt][nt] = (v8f)0.0f;

  if (tid < 32) {
    rowM[tid] = -1e30f;
    rowL[tid] = 0.0f;
  }
  __syncthreads();

  const __bf16* Qbase = Qt + ((long)b * T_N + t0) * C_N;

  for (int sb = 0; sb < T_N; sb += 128) {
    v8f sacc[2];
    sacc[0] = (v8f)0.0f;
    sacc[1] = (v8f)0.0f;
    const __bf16* Kbase = Kt + ((long)b * T_N + sb) * C_N;

    // scores: S[t,s] = sum_c Q[t,c] * K[s,c], K staged 64 channels at a time
    for (int kc2 = 0; kc2 < C_N; kc2 += 64) {
      {  // async fill: Klds[row][0..63] = K[sb+row][kc2..kc2+63]
        int row = tid >> 1;
        int h = (tid & 1) * 32;  // element offset within the 64-c chunk
        const __bf16* g = Kbase + (long)row * C_N + kc2 + h;
        async_ld16(&Klds[row * 72 + h], g);
        async_ld16(&Klds[row * 72 + h + 8], g + 8);
        async_ld16(&Klds[row * 72 + h + 16], g + 16);
        async_ld16(&Klds[row * 72 + h + 24], g + 24);
      }
      wait_async0();
      __syncthreads();
#pragma unroll
      for (int kk = 0; kk < 2; ++kk) {
        int k0 = kk * 32;
        v16bf a = load_frag(Qbase + (long)(sMt * 16) * C_N + kc2 + k0, C_N, lane);
        v16bf b0 = load_frag(&Klds[sNg * 72 + k0], 72, lane);
        v16bf b1 = load_frag(&Klds[(sNg + 16) * 72 + k0], 72, lane);
        sacc[0] = wmma_bf16(a, b0, sacc[0]);
        sacc[1] = wmma_bf16(a, b1, sacc[1]);
      }
      __syncthreads();  // before next chunk overwrites Klds
    }
#pragma unroll
    for (int j = 0; j < 2; ++j)
#pragma unroll
      for (int r = 0; r < 8; ++r)
        Sl[(sMt * 16 + hi8 + r) * 132 + sNg + j * 16 + n_l] =
            sacc[j][r] * SOFT_SCALE;
    __syncthreads();

    // online softmax, one thread per row
    if (tid < 32) {
      float mold = rowM[tid];
      float mnew = mold;
      const float* srow = &Sl[tid * 132];
      for (int i = 0; i < 128; ++i) mnew = fmaxf(mnew, srow[i]);
      float l = rowL[tid] * __expf(mold - mnew);
      __bf16* prow = &Pl[tid * 136];
      for (int i = 0; i < 128; ++i) {
        float p = __expf(srow[i] - mnew);
        l += p;
        prow[i] = f2bf(p);
      }
      rowM[tid] = mnew;
      rowL[tid] = l;
      rowScale[tid] = __expf(mold - mnew);
    }
    __syncthreads();

    // rescale O accumulators by exp(m_old - m_new)
#pragma unroll
    for (int mt = 0; mt < 2; ++mt)
#pragma unroll
      for (int r = 0; r < 8; ++r) {
        float s = rowScale[mt * 16 + hi8 + r];
#pragma unroll
        for (int nt = 0; nt < 4; ++nt) accO[mt][nt][r] *= s;
      }

    // O += P @ V^T :  O[t,c] += sum_s P[t,s] * V[c,s]
    for (int sc = 0; sc < 4; ++sc) {
      v16bf ap0 = load_frag(&Pl[0 * 136 + sc * 32], 136, lane);
      v16bf ap1 = load_frag(&Pl[16 * 136 + sc * 32], 136, lane);
      v16bf bv[4];
#pragma unroll
      for (int nt = 0; nt < 4; ++nt)
        bv[nt] = load_frag(
            V + ((long)b * C_N + wave * 64 + nt * 16) * T_N + sb + sc * 32, T_N,
            lane);
#pragma unroll
      for (int nt = 0; nt < 4; ++nt) {
        accO[0][nt] = wmma_bf16(ap0, bv[nt], accO[0][nt]);
        accO[1][nt] = wmma_bf16(ap1, bv[nt], accO[1][nt]);
      }
    }
    __syncthreads();  // Pl reads done before next iteration's Klds/Sl phase
  }

  // normalize by row sum and store O as [B,T,C] bf16
#pragma unroll
  for (int mt = 0; mt < 2; ++mt)
#pragma unroll
    for (int r = 0; r < 8; ++r) {
      int row = mt * 16 + hi8 + r;
      float inv = 1.0f / rowL[row];
      int tg = t0 + row;
#pragma unroll
      for (int nt = 0; nt < 4; ++nt) {
        int cg = wave * 64 + nt * 16 + n_l;
        Obf[((long)b * T_N + tg) * C_N + cg] = f2bf(accO[mt][nt][r] * inv);
      }
    }
}

// ---------------------------------------------------------------------------
// Kernel 3: out projection + bias + residual:
//   y[b,o,t] = sum_c Wo[o,c] * O[b,t,c] + bo[o] + x[b,o,t]
// ---------------------------------------------------------------------------
__global__ __launch_bounds__(256) void out_proj_kernel(
    const __bf16* __restrict__ Obf, const float* __restrict__ Wo,
    const float* __restrict__ bo, const float* __restrict__ x,
    float* __restrict__ y) {
  __shared__ __align__(16) __bf16 Al[128 * 40];

  const int tid = threadIdx.x;
  const int lane = tid & 31;
  const int wave = tid >> 5;
  const int waveM = wave & 1;
  const int waveN = wave >> 1;

  const int o0 = blockIdx.x * 128;
  const int t0 = blockIdx.y * 128;
  const int b = blockIdx.z;

  v8f acc[4][2];
#pragma unroll
  for (int i = 0; i < 4; ++i)
#pragma unroll
    for (int j = 0; j < 2; ++j) acc[i][j] = (v8f)0.0f;

  for (int kc = 0; kc < C_N; kc += 32) {
    {  // stage Wo tile fp32 -> bf16
      int m = tid >> 1;
      int k0 = (tid & 1) * 16;
      const float* src = Wo + (long)(o0 + m) * C_N + kc + k0;
#pragma unroll
      for (int j = 0; j < 2; ++j) {
        float4 fa = *(const float4*)(src + j * 8);
        float4 fb = *(const float4*)(src + j * 8 + 4);
        v8bf t;
        t[0] = f2bf(fa.x); t[1] = f2bf(fa.y); t[2] = f2bf(fa.z); t[3] = f2bf(fa.w);
        t[4] = f2bf(fb.x); t[5] = f2bf(fb.y); t[6] = f2bf(fb.z); t[7] = f2bf(fb.w);
        *(v8bf*)&Al[m * 40 + k0 + j * 8] = t;
      }
      if (kc + 32 < C_N) __builtin_prefetch(src + 32, 0, 0);  // next Wo tile
    }
    __syncthreads();
    v16bf af[4], bfr[2];
#pragma unroll
    for (int mt = 0; mt < 4; ++mt)
      af[mt] = load_frag(&Al[(waveM * 64 + mt * 16) * 40], 40, lane);
#pragma unroll
    for (int nt = 0; nt < 2; ++nt)
      bfr[nt] = load_frag(
          Obf + ((long)b * T_N + t0 + waveN * 32 + nt * 16) * C_N + kc, C_N, lane);
#pragma unroll
    for (int mt = 0; mt < 4; ++mt)
#pragma unroll
      for (int nt = 0; nt < 2; ++nt)
        acc[mt][nt] = wmma_bf16(af[mt], bfr[nt], acc[mt][nt]);
    __syncthreads();
  }

  const int n_l = lane & 15;
  const int hi8 = (lane >> 4) << 3;
#pragma unroll
  for (int mt = 0; mt < 4; ++mt) {
    int ob = o0 + waveM * 64 + mt * 16 + hi8;
#pragma unroll
    for (int nt = 0; nt < 2; ++nt) {
      int tg = t0 + waveN * 32 + nt * 16 + n_l;
#pragma unroll
      for (int r = 0; r < 8; ++r) {
        long idx = ((long)b * C_N + ob + r) * T_N + tg;
        y[idx] = acc[mt][nt][r] + bo[ob + r] + x[idx];
      }
    }
  }
}

extern "C" void kernel_launch(void* const* d_in, const int* in_sizes, int n_in,
                              void* d_out, int out_size, void* d_ws,
                              size_t ws_size, hipStream_t stream) {
  (void)in_sizes; (void)n_in; (void)out_size; (void)ws_size;
  const float* x = (const float*)d_in[0];
  const float* Wq = (const float*)d_in[1];
  const float* bq = (const float*)d_in[2];
  const float* Wk = (const float*)d_in[3];
  const float* bk = (const float*)d_in[4];
  const float* Wv = (const float*)d_in[5];
  const float* bv = (const float*)d_in[6];
  const float* Wo = (const float*)d_in[7];
  const float* bo = (const float*)d_in[8];
  float* y = (float*)d_out;

  const size_t n = (size_t)B_N * T_N * C_N;  // 8M elems; 4 bf16 buffers = 64 MB
  __bf16* Qt = (__bf16*)d_ws;
  __bf16* Kt = Qt + n;
  __bf16* Vb = Kt + n;
  __bf16* Obf = Vb + n;

  qkv_proj_kernel<<<dim3(4, 16, 24), 256, 0, stream>>>(x, Wq, bq, Wk, bk, Wv, bv,
                                                       Qt, Kt, Vb);
  flash_attn_kernel<<<dim3(64, 8), 256, 0, stream>>>(Qt, Kt, Vb, Obf);
  out_proj_kernel<<<dim3(4, 16, 8), 256, 0, stream>>>(Obf, Wo, bo, x, y);
}
